// MultiFieldAxialAttention_28750511079838
// MI455X (gfx1250) — matrix operand order repaired and checked by reading, not compiled, not run on hardware
//
#include <hip/hip_runtime.h>

// ---------------------------------------------------------------------------
// Types for CDNA5 WMMA (gfx1250, wave32)
// ---------------------------------------------------------------------------
typedef unsigned short u16;
typedef __attribute__((ext_vector_type(16))) __bf16       v16bf;
typedef __attribute__((ext_vector_type(8)))  float        v8f;
typedef __attribute__((ext_vector_type(4)))  unsigned int v4u;

// Exact pointee type expected by the async-LDS builtins (from hipcc diagnostic):
// "__attribute__((__vector_size__(4 * sizeof(int)))) int"
typedef int v4i __attribute__((vector_size(16)));
typedef __attribute__((address_space(1))) v4i* as1_v4i_p;   // global
typedef __attribute__((address_space(3))) v4i* as3_v4i_p;   // LDS

union FragBF {
    v16bf v;
    v4u q[2];
    unsigned int u[8];
};

// gfx1250 async global->LDS copy (ASYNCcnt-tracked, bypasses VGPRs).
// Guarded so the file still compiles if this toolchain lacks the builtin.
#if defined(__has_builtin)
#if __has_builtin(__builtin_amdgcn_global_load_async_to_lds_b128)
#define USE_ASYNC_LDS 1
#endif
#endif

__device__ __forceinline__ void cp16_g2l(const u16* __restrict__ g, u16* l) {
#ifdef USE_ASYNC_LDS
    // Flat LDS addresses carry the LDS byte offset in the low 32 bits; global
    // flat addresses are the full VA. Reinterpret into the builtin's address
    // spaces (same trick composable_kernel uses for global.load.lds).
    __builtin_amdgcn_global_load_async_to_lds_b128(
        (as1_v4i_p)(unsigned long long)g,
        (as3_v4i_p)(unsigned int)(unsigned long long)l,
        0, 0);
#else
    *(v4u*)l = *(const v4u*)g;
#endif
}

__device__ __forceinline__ void cp_async_wait() {
#ifdef USE_ASYNC_LDS
#if defined(__has_builtin) && __has_builtin(__builtin_amdgcn_s_wait_asynccnt)
    __builtin_amdgcn_s_wait_asynccnt(0);
#else
    asm volatile("s_wait_asynccnt 0x0" ::: "memory");
#endif
#endif
}

__device__ __forceinline__ u16 f32_to_bf16(float f) {
    unsigned int u = __float_as_uint(f);
    u += 0x7FFFu + ((u >> 16) & 1u);   // round-to-nearest-even
    return (u16)(u >> 16);
}

// A-fragment (16x32 bf16, M x K) from row-major [rows][ld] bf16 in LDS.
// Lanes 0-15: row = rowBase+lane, K = kBase+{0..7} (q[0]) and kBase+{16..23} (q[1])
// Lanes 16-31: same rows, K = kBase+{8..15} and kBase+{24..31}
__device__ __forceinline__ v16bf ld_fragA(const u16* s, int ld, int rowBase, int kBase, int lane) {
    int r = rowBase + (lane & 15);
    int k = kBase + ((lane >> 4) << 3);
    FragBF f;
    f.q[0] = *(const v4u*)(s + r * ld + k);
    f.q[1] = *(const v4u*)(s + r * ld + k + 16);
    return f.v;
}

// B-fragment (32x16 bf16, K x N) from row-major B^T [cols][ld] bf16 in LDS.
// Lanes 0-15: col = colBase+lane, K = kBase+{0..7} (q[0]) and kBase+{16..23} (q[1])
// Lanes 16-31: same cols,         K = kBase+{8..15} and kBase+{24..31}
__device__ __forceinline__ v16bf ld_fragB(const u16* s, int ld, int colBase, int kBase, int lane) {
    int c = colBase + (lane & 15);
    int k = kBase + ((lane >> 4) << 3);
    FragBF f;
    f.q[0] = *(const v4u*)(s + c * ld + k);
    f.q[1] = *(const v4u*)(s + c * ld + k + 16);
    return f.v;
}

// ---------------------------------------------------------------------------
// Weight cast + transpose: W [K][N] fp32 -> Wt [N][K] bf16
// ---------------------------------------------------------------------------
__global__ void wt_cast_kernel(const float* __restrict__ W, u16* __restrict__ Wt, int K, int N) {
    int idx = blockIdx.x * blockDim.x + threadIdx.x;
    if (idx < K * N) {
        int k = idx / N, n = idx - k * N;
        Wt[(size_t)n * K + k] = f32_to_bf16(W[idx]);
    }
}

// ---------------------------------------------------------------------------
// LayerNorm (no affine) fp32 -> bf16, one row per block
// ---------------------------------------------------------------------------
__global__ __launch_bounds__(256) void ln_bf16_kernel(const float* __restrict__ x,
                                                      u16* __restrict__ y, int D) {
    int row = blockIdx.x;
    const float* xr = x + (size_t)row * D;
    float s = 0.f, s2 = 0.f;
    for (int i = threadIdx.x; i < D; i += 256) { float v = xr[i]; s += v; s2 += v * v; }
    #pragma unroll
    for (int off = 16; off; off >>= 1) { s += __shfl_xor(s, off, 32); s2 += __shfl_xor(s2, off, 32); }
    __shared__ float rs[8], rs2[8];
    int w = threadIdx.x >> 5;
    if ((threadIdx.x & 31) == 0) { rs[w] = s; rs2[w] = s2; }
    __syncthreads();
    s = 0.f; s2 = 0.f;
    #pragma unroll
    for (int i = 0; i < 8; ++i) { s += rs[i]; s2 += rs2[i]; }
    float mean = s / D;
    float var = s2 / D - mean * mean;
    float r = rsqrtf(var + 1e-5f);
    for (int i = threadIdx.x; i < D; i += 256)
        y[(size_t)row * D + i] = f32_to_bf16((xr[i] - mean) * r);
}

// ---------------------------------------------------------------------------
// GEMM: C[M,N] = A[M,K](bf16) @ Bt[N,K]^T(bf16)
// Block: 256 threads (8 waves), tile 128x128, K-step 64, double-buffered LDS
// with async global->LDS copies (one s_wait_asynccnt + one barrier per step).
// Wave w: wm=w&3 -> M offset wm*32 (2 frags), wn=w>>2 -> N offset wn*64 (4 frags).
// Output: bf16 (+optional bias) OR fp32 (+bias +residual) per pointer flags.
// M,N,K are multiples of 128/128/64 for all calls here (no edge handling).
// ---------------------------------------------------------------------------
#define GBM 128
#define GBN 128
#define GBK 64

__device__ __forceinline__ void stage_tile(const u16* __restrict__ A, const u16* __restrict__ Bt,
                                           u16* dA, u16* dB, int m0, int n0, int K, int kk, int tid) {
    #pragma unroll
    for (int i = 0; i < 4; ++i) {
        int chunk = tid + i * 256;          // 0..1023
        int row = chunk >> 3;               // 8 chunks per 64-col row
        int c8 = (chunk & 7) << 3;
        cp16_g2l(&A[(size_t)(m0 + row) * K + kk + c8], &dA[row * GBK + c8]);
        cp16_g2l(&Bt[(size_t)(n0 + row) * K + kk + c8], &dB[row * GBK + c8]);
    }
}

__global__ __launch_bounds__(256)
void gemm_bf16_wmma(const u16* __restrict__ A, const u16* __restrict__ Bt,
                    u16* __restrict__ Obf, float* __restrict__ Of,
                    const float* __restrict__ bias, const float* __restrict__ resid,
                    int M, int N, int K, int ldo)
{
    __shared__ __align__(16) u16 sA[2][GBM * GBK];
    __shared__ __align__(16) u16 sB[2][GBN * GBK];

    const int tid = threadIdx.x;
    const int lane = tid & 31;
    const int w = tid >> 5;
    const int wm = w & 3;
    const int wn = w >> 2;
    const int m0 = blockIdx.y * GBM;
    const int n0 = blockIdx.x * GBN;

    v8f acc[2][4] = {};

    // Prologue: stage K-tile 0 into buffer 0.
    stage_tile(A, Bt, sA[0], sB[0], m0, n0, K, 0, tid);

    int cur = 0;
    for (int kk = 0; kk < K; kk += GBK) {
        cp_async_wait();      // own async copies for buffer `cur` have landed
        __syncthreads();      // everyone's copies visible; prior readers done
        if (kk + GBK < K)
            stage_tile(A, Bt, sA[cur ^ 1], sB[cur ^ 1], m0, n0, K, kk + GBK, tid);

        const u16* cA = sA[cur];
        const u16* cB = sB[cur];
        #pragma unroll
        for (int c = 0; c < GBK; c += 32) {
            v16bf afrag[2], bfrag[4];
            #pragma unroll
            for (int mi = 0; mi < 2; ++mi) afrag[mi] = ld_fragA(cA, GBK, wm * 32 + mi * 16, c, lane);
            #pragma unroll
            for (int ni = 0; ni < 4; ++ni) bfrag[ni] = ld_fragB(cB, GBK, wn * 64 + ni * 16, c, lane);
            #pragma unroll
            for (int mi = 0; mi < 2; ++mi)
                #pragma unroll
                for (int ni = 0; ni < 4; ++ni)
                    acc[mi][ni] = __builtin_amdgcn_wmma_f32_16x16x32_bf16(
                        false, afrag[mi], false, bfrag[ni], (short)0, acc[mi][ni], false, false);
        }
        cur ^= 1;
    }

    // Epilogue. C layout: lane n (0-15) holds col n; VGPR r holds row r (lanes<16) / r+8 (lanes>=16).
    const int ncl = lane & 15;
    const int mtop = (lane < 16) ? 0 : 8;
    #pragma unroll
    for (int mi = 0; mi < 2; ++mi) {
        #pragma unroll
        for (int ni = 0; ni < 4; ++ni) {
            int col = n0 + wn * 64 + ni * 16 + ncl;
            float bval = bias ? bias[col] : 0.0f;
            #pragma unroll
            for (int r = 0; r < 8; ++r) {
                int m = m0 + wm * 32 + mi * 16 + mtop + r;
                float vv = acc[mi][ni][r] + bval;
                if (Of) {
                    size_t o = (size_t)m * ldo + col;
                    Of[o] = vv + (resid ? resid[o] : 0.0f);
                } else {
                    Obf[(size_t)m * ldo + col] = f32_to_bf16(vv);
                }
            }
        }
    }
}

// ---------------------------------------------------------------------------
// Axial attention, one wave (32 threads) per (b, u, v, head) problem.
//   Q: [t<=12][64] rows q[(qbase+m*q_sT)*ldq + h*64 ..]
//   K: [tkv<=12][64], V: same rows, cols 512+h*64 (kv buffer holds [K|V])
//   O: rows like Q, cols ocol + 0..63 (same for every head)
// S = Q K^T * 0.125 ; P = softmax_j(S) ; O = P V.  Pads t,tkv -> 16 with zeros.
// ---------------------------------------------------------------------------
__global__ __launch_bounds__(32)
void axial_attn_kernel(const u16* __restrict__ q, const u16* __restrict__ kv,
                       u16* __restrict__ o,
                       int ldq, int ldkv, int ldo, int ocol,
                       int t, int tkv,
                       int q_sB, int q_sU, int q_sV, int q_sT,
                       int kv_sB, int kv_sU, int kv_sV, int kv_sT,
                       int du, int dv)
{
    __shared__ __align__(16) u16 sQ[16 * 64];
    __shared__ __align__(16) u16 sK[16 * 64];
    __shared__ __align__(16) u16 sVt[64 * 32];   // V transposed: [e][k], k padded to 32
    __shared__ __align__(16) u16 sP[16 * 16];

    int bid = blockIdx.x;
    const int h = bid & 7;  bid >>= 3;
    const int v = bid % 12; bid /= 12;
    const int u = bid % 12; bid /= 12;
    const int b = bid;

    const int qbase  = b * q_sB + u * q_sU + v * q_sV;
    const int kvbase = b * kv_sB + (u / du) * kv_sU + (v / dv) * kv_sV;
    const int qcol = h * 64, kcol = h * 64, vcol = 512 + h * 64;
    const int lane = threadIdx.x;

    // Stage Q and K (16B vector chunks, zero-pad rows >= t / tkv)
    for (int i = lane; i < 16 * 8; i += 32) {        // 8 chunks of 8 bf16 per 64-wide row
        int m = i >> 3, c8 = (i & 7) << 3;
        v4u qv = {}, kvv = {};
        if (m < t)   qv  = *(const v4u*)&q[(size_t)(qbase + m * q_sT) * ldq + qcol + c8];
        if (m < tkv) kvv = *(const v4u*)&kv[(size_t)(kvbase + m * kv_sT) * ldkv + kcol + c8];
        *(v4u*)&sQ[m * 64 + c8] = qv;
        *(v4u*)&sK[m * 64 + c8] = kvv;
    }
    // Stage V transposed (vector loads, scatter stores; zero-pad K dim to 32)
    for (int i = lane; i < 64 * 32 / 8; i += 32) {
        v4u z = {};
        *(v4u*)&sVt[i * 8] = z;
    }
    __syncthreads();
    for (int i = lane; i < tkv * 8; i += 32) {
        int m = i >> 3, c8 = (i & 7) << 3;
        union { v4u vv; u16 s[8]; } uu;
        uu.vv = *(const v4u*)&kv[(size_t)(kvbase + m * kv_sT) * ldkv + vcol + c8];
        #pragma unroll
        for (int j = 0; j < 8; ++j) sVt[(c8 + j) * 32 + m] = uu.s[j];
    }
    __syncthreads();

    // S = Q K^T  (M=16 rows t, N=16 cols tkv, Kdim=64 -> two WMMAs)
    v8f s_acc = {};
    #pragma unroll
    for (int c = 0; c < 64; c += 32) {
        v16bf aq = ld_fragA(sQ, 64, 0, c, lane);
        v16bf bk = ld_fragB(sK, 64, 0, c, lane);   // B = K^T, so B^T = K rows
        s_acc = __builtin_amdgcn_wmma_f32_16x16x32_bf16(false, aq, false, bk,
                                                        (short)0, s_acc, false, false);
    }

    // Row-wise softmax. Lane's column = lane&15; each VGPR r is a distinct row,
    // spread across 16 lanes of one half-wave -> xor-shuffle reductions over masks 1..8.
    const int ncl = lane & 15;
    #pragma unroll
    for (int r = 0; r < 8; ++r) {
        float sv = s_acc[r] * 0.125f;              // e^-0.5, e=64
        if (ncl >= tkv) sv = -1e30f;               // mask padded KV columns
        float mx = sv;
        #pragma unroll
        for (int off = 1; off < 16; off <<= 1) mx = fmaxf(mx, __shfl_xor(mx, off, 32));
        float ev = __expf(sv - mx);
        float sm = ev;
        #pragma unroll
        for (int off = 1; off < 16; off <<= 1) sm += __shfl_xor(sm, off, 32);
        float pr = ev / sm;
        int row = (lane < 16) ? r : r + 8;
        sP[row * 16 + ncl] = f32_to_bf16(pr);      // relayout via LDS
    }
    __syncthreads();

    // P A-fragment: real K = 16 (cols of P), pad to 32 with zeros
    FragBF pf;
    {
        int row = lane & 15;
        int k = (lane >> 4) << 3;
        pf.q[0] = *(const v4u*)(sP + row * 16 + k);
        pf.u[4] = pf.u[5] = pf.u[6] = pf.u[7] = 0u;
    }

    // O = P V : N = 64 -> four 16-wide tiles; B^T = V^T already staged in sVt
    const int mtop = (lane < 16) ? 0 : 8;
    #pragma unroll
    for (int ni = 0; ni < 4; ++ni) {
        v16bf bv = ld_fragB(sVt, 32, ni * 16, 0, lane);
        v8f oacc = {};
        oacc = __builtin_amdgcn_wmma_f32_16x16x32_bf16(false, pf.v, false, bv,
                                                       (short)0, oacc, false, false);
        #pragma unroll
        for (int r = 0; r < 8; ++r) {
            int m = mtop + r;
            if (m < t)
                o[(size_t)(qbase + m * q_sT) * ldo + ocol + ni * 16 + ncl] = f32_to_bf16(oacc[r]);
        }
    }
}

// ---------------------------------------------------------------------------
// Host orchestration
// ---------------------------------------------------------------------------
extern "C" void kernel_launch(void* const* d_in, const int* in_sizes, int n_in,
                              void* d_out, int out_size, void* d_ws, size_t ws_size,
                              hipStream_t stream)
{
    (void)in_sizes; (void)n_in; (void)out_size; (void)ws_size;
    const float* x0        = (const float*)d_in[0];
    const float* x1        = (const float*)d_in[1];
    const float* wq_self   = (const float*)d_in[2];
    const float* wkv_self  = (const float*)d_in[3];
    const float* wq_cross  = (const float*)d_in[4];
    const float* wkv_cross = (const float*)d_in[5];
    const float* w_out     = (const float*)d_in[6];
    const float* b_out     = (const float*)d_in[7];

    const int B = 32, T = 12, LA = 12, LO = 12;
    const int M0 = B * T * LA * LO;                 // 55296 (div by 128)
    const int M1 = B * T * (LA / 2) * (LO / 2);     // 13824 (div by 128)

    char* p = (char*)d_ws;
    auto carve = [&](size_t bytes) -> void* {
        void* r = (void*)p;
        p += (bytes + 255) & ~(size_t)255;
        return r;
    };
    u16* f0     = (u16*)carve((size_t)M0 * 1024 * 2);   // current field0 (bf16)
    u16* f1     = (u16*)carve((size_t)M1 * 512 * 2);    // LN(x1) (bf16)
    u16* qs     = (u16*)carve((size_t)M0 * 512 * 2);
    u16* kvs    = (u16*)carve((size_t)M0 * 1024 * 2);
    u16* qc     = (u16*)carve((size_t)M0 * 512 * 2);
    u16* kvc    = (u16*)carve((size_t)M1 * 1024 * 2);
    u16* ob     = (u16*)carve((size_t)M0 * 1024 * 2);   // [o_self | o_cross]
    u16* wqs_t  = (u16*)carve((size_t)3 * 512 * 1024 * 2);
    u16* wkvs_t = (u16*)carve((size_t)3 * 1024 * 1024 * 2);
    u16* wqc_t  = (u16*)carve((size_t)3 * 512 * 1024 * 2);
    u16* wkvc_t = (u16*)carve((size_t)3 * 1024 * 512 * 2);
    u16* wout_t = (u16*)carve((size_t)3 * 1024 * 1024 * 2);

    // One-time weight cast+transpose to bf16 [N][K]
    for (int ip = 0; ip < 3; ++ip) {
        wt_cast_kernel<<<(1024 * 512 + 255) / 256, 256, 0, stream>>>(
            wq_self + (size_t)ip * 1024 * 512, wqs_t + (size_t)ip * 512 * 1024, 1024, 512);
        wt_cast_kernel<<<(1024 * 1024 + 255) / 256, 256, 0, stream>>>(
            wkv_self + (size_t)ip * 1024 * 1024, wkvs_t + (size_t)ip * 1024 * 1024, 1024, 1024);
        wt_cast_kernel<<<(1024 * 512 + 255) / 256, 256, 0, stream>>>(
            wq_cross + (size_t)ip * 1024 * 512, wqc_t + (size_t)ip * 512 * 1024, 1024, 512);
        wt_cast_kernel<<<(512 * 1024 + 255) / 256, 256, 0, stream>>>(
            wkv_cross + (size_t)ip * 512 * 1024, wkvc_t + (size_t)ip * 1024 * 512, 512, 1024);
        wt_cast_kernel<<<(1024 * 1024 + 255) / 256, 256, 0, stream>>>(
            w_out + (size_t)ip * 1024 * 1024, wout_t + (size_t)ip * 1024 * 1024, 1024, 1024);
    }

    // LayerNorms (once)
    ln_bf16_kernel<<<M0, 256, 0, stream>>>(x0, f0, 1024);
    ln_bf16_kernel<<<M1, 256, 0, stream>>>(x1, f1, 512);

    // Per-perm strides (field0: sB=1728,sT=144,sLA=12,sLO=1; field1: 432,36,6,1)
    // perm0 seq=T, outer=(LA,LO); perm1 seq=LA, outer=(T,LO); perm2 seq=LO, outer=(T,LA)
    static const int qsU[3] = {12, 144, 144}, qsV[3] = {1, 1, 12}, qsT[3] = {144, 12, 1};
    static const int csU[3] = {6, 36, 36},    csV[3] = {1, 1, 6},  csT[3] = {36, 6, 1};
    static const int cdu[3] = {2, 1, 1},      cdv[3] = {2, 2, 2},  ctkv[3] = {12, 6, 6};

    for (int ip = 0; ip < 3; ++ip) {
        gemm_bf16_wmma<<<dim3(4, M0 / 128), 256, 0, stream>>>(
            f0, wqs_t + (size_t)ip * 512 * 1024, qs, nullptr, nullptr, nullptr, M0, 512, 1024, 512);
        gemm_bf16_wmma<<<dim3(8, M0 / 128), 256, 0, stream>>>(
            f0, wkvs_t + (size_t)ip * 1024 * 1024, kvs, nullptr, nullptr, nullptr, M0, 1024, 1024, 1024);
        gemm_bf16_wmma<<<dim3(4, M0 / 128), 256, 0, stream>>>(
            f0, wqc_t + (size_t)ip * 512 * 1024, qc, nullptr, nullptr, nullptr, M0, 512, 1024, 512);
        // cross-KV at native x1 resolution (no repeat materialization)
        gemm_bf16_wmma<<<dim3(8, M1 / 128), 256, 0, stream>>>(
            f1, wkvc_t + (size_t)ip * 1024 * 512, kvc, nullptr, nullptr, nullptr, M1, 1024, 512, 1024);

        const int nblk = B * 12 * 12 * 8;
        axial_attn_kernel<<<nblk, 32, 0, stream>>>(
            qs, kvs, ob, 512, 1024, 1024, 0, 12, 12,
            1728, qsU[ip], qsV[ip], qsT[ip],
            1728, qsU[ip], qsV[ip], qsT[ip], 1, 1);
        axial_attn_kernel<<<nblk, 32, 0, stream>>>(
            qc, kvc, ob, 512, 1024, 1024, 512, 12, ctkv[ip],
            1728, qsU[ip], qsV[ip], qsT[ip],
            432, csU[ip], csV[ip], csT[ip], cdu[ip], cdv[ip]);

        if (ip < 2) {
            gemm_bf16_wmma<<<dim3(8, M0 / 128), 256, 0, stream>>>(
                ob, wout_t + (size_t)ip * 1024 * 1024, f0, nullptr,
                b_out + (size_t)ip * 1024, nullptr, M0, 1024, 1024, 1024);
        } else {
            // final round: fp32 out = acc + bias + x0 residual, straight to d_out
            gemm_bf16_wmma<<<dim3(8, M0 / 128), 256, 0, stream>>>(
                ob, wout_t + (size_t)ip * 1024 * 1024, nullptr, (float*)d_out,
                b_out + (size_t)ip * 1024, x0, M0, 1024, 1024, 1024);
        }
    }
}
